// SelfAttention1D_68839735820511
// MI455X (gfx1250) — matrix operand order repaired and checked by reading, not compiled, not run on hardware
//
#include <hip/hip_runtime.h>
#include <hip/hip_bf16.h>

typedef __attribute__((ext_vector_type(16))) __bf16 v16bf;
typedef __attribute__((ext_vector_type(8)))  float  v8f;

#define B_  4
#define C_  1024
#define L_  2048
#define H_  8
#define HD_ 16
#define VD_ 128
#define CQ_ 128
#define KB_ 32            // keys per flash block

union Frag { v16bf v; uint4 q[2]; };

__device__ __forceinline__ unsigned short f2bf(float f) {
    unsigned u = __float_as_uint(f);
    u += 0x7fffu + ((u >> 16) & 1u);        // round-to-nearest-even
    return (unsigned short)(u >> 16);
}

// gfx1250 async copy global->LDS (ASYNCcnt-tracked), per CDNA5 ISA
__device__ __forceinline__ void async_g2l_b128(const unsigned short* g, unsigned lds_off) {
    asm volatile("global_load_async_to_lds_b128 %0, %1, off"
                 :: "v"(lds_off), "v"(g) : "memory");
}
__device__ __forceinline__ void wait_async0() {
    asm volatile("s_wait_asynccnt 0x0" ::: "memory");
}

// ---------------- stage 1a: f32 -> bf16 weight convert ----------------
__global__ void cvt_bf16_kernel(const float* __restrict__ src,
                                unsigned short* __restrict__ dst, int n) {
    int i = blockIdx.x * blockDim.x + threadIdx.x;
    int stride = gridDim.x * blockDim.x;
    for (; i < n; i += stride) dst[i] = f2bf(src[i]);
}

// ---------------- stage 1b: x(B,C,L) f32 -> xT(B,L,C) bf16 ----------------
__global__ void xpose_kernel(const float* __restrict__ x,
                             unsigned short* __restrict__ xT) {
    __shared__ float tile[32][33];
    const int b = blockIdx.z;
    const int l0 = blockIdx.x * 32, c0 = blockIdx.y * 32;
    const int tx = threadIdx.x, ty = threadIdx.y;
    #pragma unroll
    for (int i = 0; i < 32; i += 8)
        tile[ty + i][tx] = x[((size_t)b * C_ + (c0 + ty + i)) * L_ + (l0 + tx)];
    __syncthreads();
    #pragma unroll
    for (int i = 0; i < 32; i += 8)
        xT[((size_t)b * L_ + (l0 + ty + i)) * C_ + (c0 + tx)] = f2bf(tile[tx][ty + i]);
}

// ---------------- stage 2: projection GEMM (software-pipelined) -------------
// mode 0: store (B, M, L)   (for v)
// mode 1: store (B, H, L, 16) head-transposed (for q/k), with scale folded in
__global__ __launch_bounds__(128)
void proj_kernel(const unsigned short* __restrict__ Wh,
                 const unsigned short* __restrict__ xT,
                 const float* __restrict__ bias,
                 unsigned short* __restrict__ out,
                 int mode, float scale) {
    const int wave = threadIdx.x >> 5, lane = threadIdx.x & 31;
    const int lo = lane & 15, hi = lane >> 4;
    const int b = blockIdx.z;
    const int obase = (blockIdx.y * 4 + wave) * 16;
    const int nbase = blockIdx.x * 64;

    v8f acc[4] = {};
    const unsigned short* arow = Wh + (size_t)(obase + lo) * C_;
    const unsigned short* brow[4];
    #pragma unroll
    for (int nt = 0; nt < 4; ++nt)
        brow[nt] = xT + ((size_t)b * L_ + nbase + nt * 16 + lo) * C_ + 8 * hi;

    Frag A, Bf[4];
    // prologue: fragments for kk = 0
    A.q[0] = *(const uint4*)(arow + 8 * hi);
    A.q[1] = *(const uint4*)(arow + 16 + 8 * hi);
    #pragma unroll
    for (int nt = 0; nt < 4; ++nt) {
        Bf[nt].q[0] = *(const uint4*)(brow[nt]);
        Bf[nt].q[1] = *(const uint4*)(brow[nt] + 16);
    }
    for (int kk = 0; kk < C_; kk += 32) {
        // issue next-step loads before consuming current fragments
        const int kn = (kk + 32 < C_) ? kk + 32 : kk;   // stays in bounds
        Frag An, Bn[4];
        An.q[0] = *(const uint4*)(arow + kn + 8 * hi);
        An.q[1] = *(const uint4*)(arow + kn + 16 + 8 * hi);
        #pragma unroll
        for (int nt = 0; nt < 4; ++nt) {
            Bn[nt].q[0] = *(const uint4*)(brow[nt] + kn);
            Bn[nt].q[1] = *(const uint4*)(brow[nt] + kn + 16);
        }
        #pragma unroll
        for (int nt = 0; nt < 4; ++nt)
            acc[nt] = __builtin_amdgcn_wmma_f32_16x16x32_bf16(
                false, A.v, false, Bf[nt].v, (short)0, acc[nt], false, false);
        A = An;
        #pragma unroll
        for (int nt = 0; nt < 4; ++nt) Bf[nt] = Bn[nt];
    }
    #pragma unroll
    for (int i = 0; i < 8; ++i) {
        const int o = obase + i + 8 * hi;           // D: VGPR i -> row i (+8 for hi lanes)
        const float bv = bias[o];
        #pragma unroll
        for (int nt = 0; nt < 4; ++nt) {
            const int l = nbase + nt * 16 + lo;     // D: lane -> col
            const float val = (acc[nt][i] + bv) * scale;
            size_t idx;
            if (mode == 0) {
                idx = ((size_t)b * C_ + o) * L_ + l;
            } else {
                const int h = o >> 4, d = o & 15;
                idx = (((size_t)b * H_ + h) * L_ + l) * (size_t)HD_ + d;
            }
            out[idx] = f2bf(val);
        }
    }
}

// ---------------- stage 3: fused flash attention + residual ----------------
// 4 waves/WG share one (b,h); K/V tiles double-buffered in LDS via async copy.
__global__ __launch_bounds__(128)
void attn_kernel(const unsigned short* __restrict__ qT,   // (B,H,L,16) bf16, pre-scaled
                 const unsigned short* __restrict__ kT,   // (B,H,L,16) bf16
                 const unsigned short* __restrict__ vh,   // (B,C,L)    bf16
                 const float* __restrict__ x,
                 const float* __restrict__ gamma,
                 float* __restrict__ out) {
    __shared__ uint4 kbuf4[2][KB_ * HD_ / 8];       // 2 x 1 KB  : K tile (m, d)
    __shared__ uint4 vbuf4[2][VD_ * KB_ / 8];       // 2 x 8 KB  : V tile (e, m)
    __shared__ uint4 smem4[4][64];                  // 1 KB private per wave (P / transpose)

    const int tid = threadIdx.x;
    const int wave = tid >> 5, lane = tid & 31;
    const int lo = lane & 15, hi = lane >> 4;
    unsigned short* smP = (unsigned short*)smem4[wave];
    float*          smF = (float*)smem4[wave];

    const int gw   = blockIdx.x * 4 + wave;
    const int lblk = gw & (L_ / 16 - 1);
    const int bh   = gw / (L_ / 16);                // identical for all 4 waves
    const int b    = bh >> 3, h = bh & 7;
    const float g  = gamma[0];
    const uint4 Z4 = make_uint4(0, 0, 0, 0);

    const unsigned short* kbase = kT + (size_t)bh * L_ * HD_;
    const unsigned short* vbase = vh + ((size_t)b * C_ + h * VD_) * L_;

    // cooperative async stage of one 32-key K/V block into LDS buffer `buf`
    auto stage = [&](int buf, int mb) {
        if (tid < 64) {                              // K: 32 rows x 32B = 64 chunks
            const int r = tid >> 1, c = tid & 1;
            async_g2l_b128(kbase + (size_t)(mb + r) * HD_ + c * 8,
                           (unsigned)(size_t)((unsigned short*)kbuf4[buf] + r * HD_ + c * 8));
        }
        #pragma unroll
        for (int i = 0; i < 4; ++i) {                // V: 128 rows x 64B = 512 chunks
            const int chunk = i * 128 + tid;
            const int e = chunk >> 2, c = chunk & 3;
            async_g2l_b128(vbase + (size_t)e * L_ + mb + c * 8,
                           (unsigned)(size_t)((unsigned short*)vbuf4[buf] + e * KB_ + c * 8));
        }
    };

    Frag Aq;  // q block as A 16x32 (K padded 16->32 with zeros in V4-7)
    {
        const unsigned short* qp =
            qT + ((size_t)bh * L_ + lblk * 16 + lo) * HD_ + 8 * hi;
        Aq.q[0] = *(const uint4*)qp;
        Aq.q[1] = Z4;
    }

    v8f acc[8] = {};
    float mrow[8], rrow[8];
    #pragma unroll
    for (int i = 0; i < 8; ++i) { mrow[i] = -1e30f; rrow[i] = 0.f; }

    stage(0, 0);
    wait_async0();
    __syncthreads();

    const int NB = L_ / KB_;
    for (int jb = 0; jb < NB; ++jb) {
        const int cur = jb & 1;
        if (jb + 1 < NB) stage(cur ^ 1, (jb + 1) * KB_);   // prefetch next block

        const unsigned short* kb = (const unsigned short*)kbuf4[cur];
        const unsigned short* vb = (const unsigned short*)vbuf4[cur];

        v8f s[2];
        #pragma unroll
        for (int t = 0; t < 2; ++t) {   // scores: S[l][m] = sum_d q[l,d] k[m,d]
            Frag Bk;                    // B[k=d][n=m]: d = 8*hi + s in V0-3, zero K-pad in V4-7
            {
                const unsigned short* kp = kb + (t * 16 + lo) * HD_ + 8 * hi;
                Bk.q[0] = *(const uint4*)kp;
                Bk.q[1] = Z4;
            }
            v8f z = {};
            s[t] = __builtin_amdgcn_wmma_f32_16x16x32_bf16(
                false, Aq.v, false, Bk.v, (short)0, z, false, false);
        }
        // online softmax; row for VGPR i is (i + 8*hi), data across the 16-lane group
        #pragma unroll
        for (int i = 0; i < 8; ++i) {
            float bm = fmaxf(s[0][i], s[1][i]);
            #pragma unroll
            for (int off = 1; off < 16; off <<= 1)
                bm = fmaxf(bm, __shfl_xor(bm, off, 32));
            const float mnew  = fmaxf(mrow[i], bm);
            const float alpha = __expf(mrow[i] - mnew);
            const float p0 = __expf(s[0][i] - mnew);
            const float p1 = __expf(s[1][i] - mnew);
            float ps = p0 + p1;
            #pragma unroll
            for (int off = 1; off < 16; off <<= 1)
                ps += __shfl_xor(ps, off, 32);
            rrow[i] = rrow[i] * alpha + ps;
            mrow[i] = mnew;
            const int row = i + 8 * hi;
            smP[row * 32 + lo]      = f2bf(p0);
            smP[row * 32 + 16 + lo] = f2bf(p1);
            #pragma unroll
            for (int t = 0; t < 8; ++t) acc[t][i] *= alpha;
        }
        asm volatile("" ::: "memory");   // same-wave DS ops are HW-ordered
        Frag P;                          // reload probs as A-fragment (16 x 32)
        P.q[0] = *(const uint4*)(smP + lo * 32 + 8 * hi);
        P.q[1] = *(const uint4*)(smP + lo * 32 + 16 + 8 * hi);
        asm volatile("" ::: "memory");
        #pragma unroll
        for (int t = 0; t < 8; ++t) {    // out[l][e] += P[l][m] * v[e][m]
            const unsigned short* vp = vb + (t * 16 + lo) * KB_ + 8 * hi;
            Frag Bv;                     // B[k=m][n=e]: m = 8*hi+s, then 16+8*hi+s per lane
            Bv.q[0] = *(const uint4*)vp;
            Bv.q[1] = *(const uint4*)(vp + 16);
            acc[t] = __builtin_amdgcn_wmma_f32_16x16x32_bf16(
                false, P.v, false, Bv.v, (short)0, acc[t], false, false);
        }
        wait_async0();     // next buffer landed (this wave's async issues)
        __syncthreads();   // all waves done reading cur + all copies visible
    }
    #pragma unroll
    for (int i = 0; i < 8; ++i) {
        const float inv = 1.f / rrow[i];
        #pragma unroll
        for (int t = 0; t < 8; ++t) acc[t][i] *= inv;
    }
    // epilogue: transpose each 16x16 tile through LDS -> coalesced residual store
    const int lgbase = lblk * 16;
    for (int t = 0; t < 8; ++t) {
        asm volatile("" ::: "memory");
        #pragma unroll
        for (int i = 0; i < 8; ++i)
            smF[lo * 16 + (i + 8 * hi)] = acc[t][i];   // smF[e_local][l_local]
        asm volatile("" ::: "memory");
        #pragma unroll
        for (int p = 0; p < 8; ++p) {
            const int el = 2 * p + hi;
            const int cg = h * VD_ + t * 16 + el;
            const size_t idx = ((size_t)b * C_ + cg) * L_ + lgbase + lo;
            out[idx] = x[idx] + g * smF[el * 16 + lo];
        }
    }
}

// ---------------------------------------------------------------------------
extern "C" void kernel_launch(void* const* d_in, const int* in_sizes, int n_in,
                              void* d_out, int out_size, void* d_ws, size_t ws_size,
                              hipStream_t stream) {
    (void)in_sizes; (void)n_in; (void)out_size; (void)ws_size;
    const float* x     = (const float*)d_in[0];
    const float* Wq    = (const float*)d_in[1];
    const float* bq    = (const float*)d_in[2];
    const float* Wk    = (const float*)d_in[3];
    const float* bk    = (const float*)d_in[4];
    const float* Wv    = (const float*)d_in[5];
    const float* bv    = (const float*)d_in[6];
    const float* gamma = (const float*)d_in[7];
    float* out = (float*)d_out;

    char* ws = (char*)d_ws;
    size_t off = 0;
    unsigned short* xT  = (unsigned short*)(ws + off); off += (size_t)B_ * L_ * C_ * 2;
    unsigned short* Wqh = (unsigned short*)(ws + off); off += (size_t)CQ_ * C_ * 2;
    unsigned short* Wkh = (unsigned short*)(ws + off); off += (size_t)CQ_ * C_ * 2;
    unsigned short* Wvh = (unsigned short*)(ws + off); off += (size_t)C_ * C_ * 2;
    unsigned short* qT  = (unsigned short*)(ws + off); off += (size_t)B_ * CQ_ * L_ * 2;
    unsigned short* kT  = (unsigned short*)(ws + off); off += (size_t)B_ * CQ_ * L_ * 2;
    unsigned short* vh  = (unsigned short*)(ws + off); off += (size_t)B_ * C_ * L_ * 2;

    // stage 1: precision convert + transpose
    cvt_bf16_kernel<<<dim3((CQ_ * C_ + 255) / 256), dim3(256), 0, stream>>>(Wq, Wqh, CQ_ * C_);
    cvt_bf16_kernel<<<dim3((CQ_ * C_ + 255) / 256), dim3(256), 0, stream>>>(Wk, Wkh, CQ_ * C_);
    cvt_bf16_kernel<<<dim3((C_ * C_ + 255) / 256),  dim3(256), 0, stream>>>(Wv, Wvh, C_ * C_);
    xpose_kernel<<<dim3(L_ / 32, C_ / 32, B_), dim3(32, 8), 0, stream>>>(x, xT);

    // stage 2: projections (scale 1/sqrt(hd)=0.25 folded into q)
    proj_kernel<<<dim3(L_ / 64, CQ_ / 64, B_), dim3(128), 0, stream>>>(Wqh, xT, bq, qT, 1, 0.25f);
    proj_kernel<<<dim3(L_ / 64, CQ_ / 64, B_), dim3(128), 0, stream>>>(Wkh, xT, bk, kT, 1, 1.0f);
    proj_kernel<<<dim3(L_ / 64, C_ / 64, B_),  dim3(128), 0, stream>>>(Wvh, xT, bv, vh, 0, 1.0f);

    // stage 3: fused flash attention + residual
    attn_kernel<<<dim3(B_ * H_ * (L_ / 16) / 4), dim3(128), 0, stream>>>(
        qT, kT, vh, x, gamma, out);
}
